// GraphFilter_62895501082687
// MI455X (gfx1250) — matrix operand order repaired and checked by reading, not compiled, not run on hardware
//
#include <hip/hip_runtime.h>

// GraphFilter: y[f,n] = sum_{k,g} W[f,k,g] * (x^T S^k)[g,n] + b[f]
// N=8192, G=F=128, K=4 taps, E=1.
//
// MI455X plan (reasoned, compile-only):
//  * Convert S (256MB f32) to f16 **transposed** once -> 128MB, fits 192MB L2.
//    All 3 hop GEMMs then stream S out of L2; HBM floor ~384MB ~ 16us @23.3TB/s.
//  * Hops via v_wmma_f32_16x16x32_f16 (f32 accumulate). f16 keeps 3 more
//    mantissa bits than bf16; values are O(1) since S ~ 1/sqrt(N) scaled.
//  * S^T layout makes every WMMA B-fragment two contiguous 16B per-lane loads
//    (ISA 16-bit B 32x16 layout); Z row-major makes A-fragments two contiguous
//    16B chunks (ISA 16-bit A 16x32 layout). No LDS transpose in the hot loop.
//  * Each hop also emits a transposed copy ZT[n][k*128+g] (one 16B store per
//    accumulator tile per lane) so the final combine GEMM
//    y = Wf16[128x512] @ ZT^T + bias is also pure contiguous WMMA fragments.
//  * Kernel-launch ordering on `stream` provides the hop-to-hop global sync.

typedef __attribute__((ext_vector_type(16))) _Float16     v16h;
typedef __attribute__((ext_vector_type(8)))  float        v8f;
typedef __attribute__((ext_vector_type(8)))  unsigned int v8u;

static constexpr int    NN = 8192;                 // graph nodes
static constexpr int    GG = 128;                  // input features
static constexpr int    FF = 128;                  // output features
static constexpr int    KT = 4;                    // filter taps
static constexpr int    KK = KT * GG;              // 512 = stacked K dim
static constexpr size_t ZLEVEL   = (size_t)GG * NN;        // elements / Z level
static constexpr size_t S_BYTES  = (size_t)NN * NN * 2;    // f16 S^T  (128 MB)
static constexpr size_t ZRM_BYTES = (size_t)KT * ZLEVEL * 2; // 8 MB
static constexpr size_t ZT_BYTES  = (size_t)NN * KK * 2;     // 8 MB

__device__ __forceinline__ unsigned short f2h(float f) {
  _Float16 h = (_Float16)f;
  return __builtin_bit_cast(unsigned short, h);
}

// Pack two 16B LDS/global chunks into a 16xf16 A fragment
// (ISA 16-bit A 16x32: lane half picks K = half*8+{0..7} then 16+half*8+{0..7}).
__device__ __forceinline__ v16h load_a_frag(const unsigned short* p) {
  uint4 lo = *(const uint4*)p;          // elements 0..7
  uint4 hi = *(const uint4*)(p + 16);   // elements 8..15 (K+16 block)
  v8u t;
  t[0] = lo.x; t[1] = lo.y; t[2] = lo.z; t[3] = lo.w;
  t[4] = hi.x; t[5] = hi.y; t[6] = hi.z; t[7] = hi.w;
  return __builtin_bit_cast(v16h, t);
}

// ---------------------------------------------------------------------------
// Prep kernels
// ---------------------------------------------------------------------------

// S^T in f16: STh[n*NN + m] = (f16)S[m*NN + n]. LDS 32x32 tiled transpose,
// coalesced reads (f32) and writes (f16). NT read: keep f32 S out of L2 so
// the f16 copy owns the cache.
__global__ __launch_bounds__(256)
void s_transpose_f16_kernel(const float* __restrict__ S,
                            unsigned short* __restrict__ STh) {
  __shared__ float tile[32][33];
  const int t = threadIdx.x, c = t & 31, r0 = t >> 5;
  const size_t n0 = (size_t)blockIdx.x * 32;
  const size_t m0 = (size_t)blockIdx.y * 32;
#pragma unroll
  for (int rr = r0; rr < 32; rr += 8)
    tile[rr][c] = __builtin_nontemporal_load(S + (m0 + rr) * NN + n0 + c);
  __syncthreads();
#pragma unroll
  for (int rr = r0; rr < 32; rr += 8)
    STh[(n0 + rr) * NN + m0 + c] = f2h(tile[c][rr]);
}

// ZT[:, 0:128] = x itself (ZT[n][g] = x[n][g]), plain f32->f16 stream.
__global__ __launch_bounds__(256)
void x_to_zt_kernel(const float* __restrict__ x,
                    unsigned short* __restrict__ zt) {
  const int i = blockIdx.x * 256 + threadIdx.x;  // < NN*GG
  const int n = i >> 7, g = i & 127;
  zt[(size_t)n * KK + g] = f2h(x[i]);
}

// Zrm0 = x^T (row-major [G][N]) via LDS tiled transpose.
__global__ __launch_bounds__(256)
void x_to_zrm_kernel(const float* __restrict__ x,
                     unsigned short* __restrict__ zrm) {
  __shared__ float tile[32][33];
  const int t = threadIdx.x, c = t & 31, r0 = t >> 5;
  const int n0 = blockIdx.x * 32, g0 = blockIdx.y * 32;
#pragma unroll
  for (int rr = r0; rr < 32; rr += 8)
    tile[rr][c] = x[(size_t)(n0 + rr) * GG + g0 + c];
  __syncthreads();
#pragma unroll
  for (int rr = r0; rr < 32; rr += 8)
    zrm[(size_t)(g0 + rr) * NN + n0 + c] = f2h(tile[c][rr]);
}

// Wh[f][k*128+g] = (f16)weight[f][0][k][g]  (flat identity copy, E=1).
__global__ __launch_bounds__(256)
void w_convert_kernel(const float* __restrict__ w,
                      unsigned short* __restrict__ wh) {
  const int i = blockIdx.x * 256 + threadIdx.x;  // < FF*KK
  wh[i] = f2h(w[i]);
}

// ---------------------------------------------------------------------------
// WMMA GEMM: C[M x 8192] = A[M x Kdim] * B, with B supplied transposed
// (BT[n][k] = B[k][n]) so B fragments are two contiguous 16B chunks per lane.
// Block = 128 threads = 4 waves (2x2); wave tile 32x64; block tile 64x128.
// COMBINE=false: hop. Writes f16 row-major (next A) + transposed ZT slice.
// COMBINE=true : writes f32 output + bias.
// ---------------------------------------------------------------------------
template <bool COMBINE>
__global__ __launch_bounds__(128)
void wmma_gemm_kernel(const unsigned short* __restrict__ A, int lda,
                      const unsigned short* __restrict__ BT, int ldbt,
                      int Kdim,
                      unsigned short* __restrict__ zrmOut,
                      unsigned short* __restrict__ ztOut,
                      int ztLd, int ztColBase,
                      float* __restrict__ outF,
                      const float* __restrict__ bias) {
  const int t    = threadIdx.x;
  const int lane = t & 31;
  const int wave = t >> 5;
  const int wm   = wave >> 1;      // 0..1
  const int wn   = wave & 1;       // 0..1
  const int half = lane >> 4;      // 0..1  (K-half select)
  const int l16  = lane & 15;

  const int m0  = blockIdx.y * 64;
  const int n0  = blockIdx.x * 128;
  const int m_w = m0 + wm * 32;
  const int n_w = n0 + wn * 64;

  const v8f vz = {0.f, 0.f, 0.f, 0.f, 0.f, 0.f, 0.f, 0.f};
  v8f acc[2][4];
#pragma unroll
  for (int i = 0; i < 2; ++i)
#pragma unroll
    for (int j = 0; j < 4; ++j) acc[i][j] = vz;

#pragma unroll 2
  for (int k0 = 0; k0 < Kdim; k0 += 32) {
    v16h af[2];
    v16h bf[4];
#pragma unroll
    for (int i = 0; i < 2; ++i)
      af[i] = load_a_frag(A + (size_t)(m_w + 16 * i + l16) * lda + k0 + half * 8);
#pragma unroll
    for (int j = 0; j < 4; ++j)   // B 32x16: lane half picks K = half*8+{0..7} then 16+half*8+{0..7}
      bf[j] = load_a_frag(BT + (size_t)(n_w + 16 * j + l16) * ldbt + k0 + half * 8);
#pragma unroll
    for (int i = 0; i < 2; ++i)
#pragma unroll
      for (int j = 0; j < 4; ++j)
        acc[i][j] = __builtin_amdgcn_wmma_f32_16x16x32_f16(
            false, af[i], false, bf[j], (short)0, acc[i][j], false, false);
  }

  // C/D 16x16 f32: vgpr r, lanes 0-15 -> M=r, lanes 16-31 -> M=8+r; N=lane%16.
#pragma unroll
  for (int i = 0; i < 2; ++i) {
#pragma unroll
    for (int j = 0; j < 4; ++j) {
      const int gRow0 = m_w + 16 * i + 8 * half;
      const int gCol  = n_w + 16 * j + l16;
      if (COMBINE) {
#pragma unroll
        for (int r = 0; r < 8; ++r)
          outF[(size_t)(gRow0 + r) * NN + gCol] = acc[i][j][r] + bias[gRow0 + r];
      } else {
        union { unsigned short u[8]; uint4 v; } pk;
#pragma unroll
        for (int r = 0; r < 8; ++r) {
          const unsigned short h = f2h(acc[i][j][r]);
          zrmOut[(size_t)(gRow0 + r) * NN + gCol] = h;  // row-major, next hop A
          pk.u[r] = h;
        }
        // transposed copy: 8 consecutive g's per lane -> one 16B store
        *(uint4*)(ztOut + (size_t)gCol * ztLd + ztColBase + gRow0) = pk.v;
      }
    }
  }
}

// ---------------------------------------------------------------------------
extern "C" void kernel_launch(void* const* d_in, const int* in_sizes, int n_in,
                              void* d_out, int out_size, void* d_ws, size_t ws_size,
                              hipStream_t stream) {
  const float* x    = (const float*)d_in[0];   // [8192,128] f32
  const float* S    = (const float*)d_in[1];   // [1,8192,8192] f32
  const float* W    = (const float*)d_in[2];   // [128,1,4,128] f32
  const float* bias = (const float*)d_in[3];   // [128,1] f32
  // d_in[4] = ind (unused by reference)
  float* out = (float*)d_out;                  // [1,128,8192] f32

  // Workspace layout (needs ~151 MB):
  char* ws = (char*)d_ws;
  unsigned short* STh = (unsigned short*)ws;                               // S^T f16
  unsigned short* Zrm = (unsigned short*)(ws + S_BYTES);                   // 4x[128x8192] f16
  unsigned short* ZT  = (unsigned short*)(ws + S_BYTES + ZRM_BYTES);       // [8192x512] f16
  unsigned short* Wh  = (unsigned short*)(ws + S_BYTES + ZRM_BYTES + ZT_BYTES); // [128x512] f16

  // 1) prep: S^T f16 (L2-resident afterwards), z0 in both layouts, W f16
  s_transpose_f16_kernel<<<dim3(NN / 32, NN / 32), 256, 0, stream>>>(S, STh);
  x_to_zt_kernel<<<(NN * GG) / 256, 256, 0, stream>>>(x, ZT);
  x_to_zrm_kernel<<<dim3(NN / 32, GG / 32), 256, 0, stream>>>(x, Zrm);
  w_convert_kernel<<<(FF * KK) / 256, 256, 0, stream>>>(W, Wh);

  // 2) hops: Z_{k+1} = Z_k @ S   ([128x8192]@[8192x8192], WMMA f16)
  for (int k = 0; k < KT - 1; ++k) {
    wmma_gemm_kernel<false><<<dim3(NN / 128, GG / 64), 128, 0, stream>>>(
        Zrm + (size_t)k * ZLEVEL, NN, STh, NN, NN,
        Zrm + (size_t)(k + 1) * ZLEVEL, ZT, KK, (k + 1) * GG,
        nullptr, nullptr);
  }

  // 3) combine: y = Wh[128x512] @ ZT^T + bias  (f32 out)
  wmma_gemm_kernel<true><<<dim3(NN / 128, FF / 64), 128, 0, stream>>>(
      Wh, KK, ZT, KK, KK,
      nullptr, nullptr, 0, 0,
      out, bias);
}
